// _RPN_52398601011253
// MI455X (gfx1250) — matrix operand order repaired and checked by reading, not compiled, not run on hardware
//
#include <hip/hip_runtime.h>
#include <cstdint>

typedef __attribute__((ext_vector_type(16))) __bf16 v16bf;
typedef __attribute__((ext_vector_type(8)))  float  v8f;

#define N_B    4
#define N_HW   64
#define N_HWP  66                          // padded (1-px halo)
#define N_CIN  512
#define N_PIX  (N_B * N_HW * N_HW)         // 16384
#define N_ANCH 9
#define N_ALL  (N_HW * N_HW * N_ANCH)      // 36864 per image
#define PRE_TOPN 6000
#define POST_N   300
#define NMS_TH   0.7f
#define NHEAD    96                        // 18 cls + 36 bbox + 4*8 heads + 10 pad

// ---- output offsets (floats) in d_out, return-tuple order ----
#define OFF_ROIS 0
#define OFF_CLS  (N_B * POST_N * 5)                 // 6000
#define OFF_DW   (OFF_CLS + N_B * 18 * 4096)        // 300912
#define OFF_DH   (OFF_DW  + N_B * 8 * 4096)         // 431984
#define OFF_BW   (OFF_DH  + N_B * 8 * 4096)         // 563056
#define OFF_BH   (OFF_BW  + N_B * 8 * 4096)         // 694128

// ---- CDNA5 async global->LDS copies (ASYNCcnt-tracked) ----
__device__ __forceinline__ void async_to_lds_b64(unsigned lds_off, const void* g) {
  asm volatile("global_load_async_to_lds_b64 %0, %1, off" :: "v"(lds_off), "v"(g) : "memory");
}
__device__ __forceinline__ void async_to_lds_b128(unsigned lds_off, const void* g) {
  asm volatile("global_load_async_to_lds_b128 %0, %1, off" :: "v"(lds_off), "v"(g) : "memory");
}
__device__ __forceinline__ void wait_async_3() {
#if __has_builtin(__builtin_amdgcn_s_wait_asynccnt)
  __builtin_amdgcn_s_wait_asynccnt(3);
#else
  asm volatile("s_wait_asynccnt 0x3" ::: "memory");
#endif
}
__device__ __forceinline__ void wait_async_0() {
#if __has_builtin(__builtin_amdgcn_s_wait_asynccnt)
  __builtin_amdgcn_s_wait_asynccnt(0);
#else
  asm volatile("s_wait_asynccnt 0x0" ::: "memory");
#endif
}

// ============================================================
// K0: NCHW f32 -> padded NHWC bf16 [4][66][66][512] (halo = 0)
// ============================================================
__global__ __launch_bounds__(256) void k_nchw_to_nhwc_pad(const float* __restrict__ in,
                                                          __bf16* __restrict__ out) {
  size_t i = (size_t)blockIdx.x * 256 + threadIdx.x;
  if (i >= (size_t)N_B * N_HWP * N_HWP * N_CIN) return;
  int c = (int)(i & 511);
  size_t sp3 = i >> 9;
  int xp = (int)(sp3 % N_HWP);
  size_t t = sp3 / N_HWP;
  int yp = (int)(t % N_HWP);
  int b  = (int)(t / N_HWP);
  float v = 0.f;
  if (xp >= 1 && xp <= N_HW && yp >= 1 && yp <= N_HW) {
    v = in[((size_t)(b * N_CIN + c) * 4096) + (size_t)(yp - 1) * N_HW + (xp - 1)];
  }
  out[i] = (__bf16)v;
}

// ============================================================
// K1a: conv_w [co][ci][3][3] f32 -> W2 [tap][co][ci] bf16
// ============================================================
__global__ __launch_bounds__(256) void k_pack_w2(const float* __restrict__ w,
                                                 __bf16* __restrict__ W2) {
  size_t i = (size_t)blockIdx.x * 256 + threadIdx.x;
  if (i >= (size_t)9 * 512 * 512) return;
  int ci = (int)(i & 511);
  size_t t2 = i >> 9;
  int co = (int)(t2 & 511);
  int tp = (int)(t2 >> 9);
  W2[i] = (__bf16)w[((size_t)co * 512 + ci) * 9 + tp];
}

// ============================================================
// K1b: fuse six 1x1 heads into Wh[96][512] bf16 + biasH[96]
// ============================================================
__global__ __launch_bounds__(256) void k_pack_head(
    const float* cls_w, const float* cls_b, const float* bbox_w, const float* bbox_b,
    const float* dw_w, const float* dw_b, const float* dh_w, const float* dh_b,
    const float* bw_w, const float* bw_b, const float* bh_w, const float* bh_b,
    __bf16* __restrict__ Wh, float* __restrict__ biasH) {
  int i = blockIdx.x * 256 + threadIdx.x;
  if (i < NHEAD * 512) {
    int ci = i & 511, co = i >> 9;
    float v = 0.f;
    if      (co < 18) v = cls_w[co * 512 + ci];
    else if (co < 54) v = bbox_w[(co - 18) * 512 + ci];
    else if (co < 62) v = dw_w[(co - 54) * 512 + ci];
    else if (co < 70) v = dh_w[(co - 62) * 512 + ci];
    else if (co < 78) v = bw_w[(co - 70) * 512 + ci];
    else if (co < 86) v = bh_w[(co - 78) * 512 + ci];
    Wh[i] = (__bf16)v;
  } else if (i < NHEAD * 512 + NHEAD) {
    int co = i - NHEAD * 512;
    float v = 0.f;
    if      (co < 18) v = cls_b[co];
    else if (co < 54) v = bbox_b[co - 18];
    else if (co < 62) v = dw_b[co - 54];
    else if (co < 70) v = dh_b[co - 62];
    else if (co < 78) v = bw_b[co - 70];
    else if (co < 86) v = bh_b[co - 78];
    biasH[co] = v;
  }
}

// ============================================================
// K2: 3x3 conv (pad 1) as implicit GEMM, bf16 WMMA, f32 acc.
// Block: 256 thr (8 waves). Tile: 32 pixels x 128 out-channels.
// Double-buffered LDS filled with GLOBAL_LOAD_ASYNC_TO_LDS.
// Dynamic LDS layout (base offset 0, no static LDS in kernel):
//   A: [2][32 rows][40 bf16]  @ 0      (buf stride 2560 B)
//   B: [2][128 rows][40 bf16] @ 5120   (buf stride 10240 B)
// grid = (512 M-tiles, 4 N-blocks), dynShared = 25600 B
// ============================================================
#define A_BUF_STRIDE 2560
#define B_BASE       5120
#define B_BUF_STRIDE 10240
#define N_STAGE      144     // 9 taps * (512/32)

__global__ __launch_bounds__(256) void k_conv3x3(const __bf16* __restrict__ F,
                                                 const __bf16* __restrict__ W2,
                                                 const float* __restrict__ bias,
                                                 __bf16* __restrict__ C1) {
  extern __shared__ char smem[];
  const int tid  = threadIdx.x;
  const int wave = tid >> 5, lane = tid & 31;
  const int pbase = blockIdx.x * 32;
  const int b   = pbase >> 12;
  const int rem = pbase & 4095;
  const int y   = rem >> 6;
  const int x0  = rem & 63;
  const int cob = blockIdx.y * 128;

  // per-thread async-copy assignments (uniform: 3 ops/thread/stage)
  const int am  = tid >> 3;          // A row 0..31
  const int ac4 = (tid & 7) * 4;     // A ci sub-offset (4 bf16 = 8 B)
  const int br0 = tid >> 2;          // B row 0..63 (second chunk: +64)
  const int bc8 = (tid & 3) * 8;     // B ci sub-offset (8 bf16 = 16 B)
  const unsigned a_dst0 = (unsigned)(am * 80 + ac4 * 2);
  const unsigned b_dst0 = (unsigned)(B_BASE + br0 * 80 + bc8 * 2);

  v8f acc0 = {};
  v8f acc1 = {};

  auto issue = [&](int ss, int buf) {
    const int tp = ss >> 4;
    const int cb = (ss & 15) << 5;
    const int dy = tp / 3 - 1, dx = tp % 3 - 1;
    const __bf16* ga = F + ((size_t)((b * N_HWP + y + 1 + dy) * N_HWP) +
                            (x0 + 1 + am + dx)) * N_CIN + cb + ac4;
    async_to_lds_b64(a_dst0 + (unsigned)buf * A_BUF_STRIDE, ga);
    const __bf16* gb0 = W2 + (((size_t)tp * 512 + cob + br0) * 512 + cb + bc8);
    const unsigned lb0 = b_dst0 + (unsigned)buf * B_BUF_STRIDE;
    async_to_lds_b128(lb0, gb0);
    async_to_lds_b128(lb0 + 64u * 80u, gb0 + 64 * 512);
  };

  issue(0, 0);
  for (int ss = 0; ss < N_STAGE; ++ss) {
    const int buf = ss & 1;
    if (ss + 1 < N_STAGE) {
      issue(ss + 1, buf ^ 1);
      wait_async_3();   // oldest 3 (stage ss) landed in LDS
    } else {
      wait_async_0();
    }
    __syncthreads();

    // ---- fragments per ISA 7.12.2 lane layout ----
    const __bf16* As = (const __bf16*)(smem) + buf * (A_BUF_STRIDE / 2);
    const __bf16* Bs = (const __bf16*)(smem + B_BASE) + buf * (B_BUF_STRIDE / 2);
    const int ln = lane & 15;
    const int k0 = (lane < 16) ? 0 : 8;
    const int kb = (lane < 16) ? 0 : 16;
    const __bf16* ar0 = As + ln * 40;
    const __bf16* ar1 = As + (16 + ln) * 40;
    const __bf16* brr = Bs + (wave * 16 + ln) * 40;
    v16bf a0, a1, bb;
    #pragma unroll
    for (int i = 0; i < 8; ++i) {
      a0[i]     = ar0[k0 + i];
      a0[8 + i] = ar0[16 + k0 + i];
      a1[i]     = ar1[k0 + i];
      a1[8 + i] = ar1[16 + k0 + i];
    }
    #pragma unroll
    for (int i = 0; i < 8; ++i) { bb[i] = brr[k0 + i]; bb[8 + i] = brr[16 + k0 + i]; }

    acc0 = __builtin_amdgcn_wmma_f32_16x16x32_bf16(false, a0, false, bb,
                                                   (short)0, acc0, false, false);
    acc1 = __builtin_amdgcn_wmma_f32_16x16x32_bf16(false, a1, false, bb,
                                                   (short)0, acc1, false, false);
    __syncthreads();   // protect buf before stage ss+2 overwrites it
  }

  // epilogue: bias + ReLU -> NHWC bf16
  const int co  = cob + wave * 16 + (lane & 15);
  const float bsv = bias[co];
  const int mh = (lane < 16) ? 0 : 8;
  #pragma unroll
  for (int r = 0; r < 8; ++r) {
    float v0 = acc0[r] + bsv;
    float v1 = acc1[r] + bsv;
    v0 = v0 > 0.f ? v0 : 0.f;
    v1 = v1 > 0.f ? v1 : 0.f;
    C1[(size_t)(pbase + r + mh) * N_CIN + co]      = (__bf16)v0;
    C1[(size_t)(pbase + 16 + r + mh) * N_CIN + co] = (__bf16)v1;
  }
}

// ============================================================
// K3: fused 1x1 heads GEMM: M=16 pixels/block, N=96, K=512
// waves 0..5 compute; all 8 waves help stage.
// ============================================================
__global__ __launch_bounds__(256) void k_head1x1(const __bf16* __restrict__ C1,
                                                 const __bf16* __restrict__ Wh,
                                                 const float* __restrict__ biasH,
                                                 float* __restrict__ S) {
  __shared__ alignas(16) __bf16 Alds[16][40];
  __shared__ alignas(16) __bf16 Blds[NHEAD][40];
  const int tid = threadIdx.x;
  const int wave = tid >> 5, lane = tid & 31;
  const int pbase = blockIdx.x * 16;

  v8f acc = {};

  for (int cb = 0; cb < N_CIN; cb += 32) {
    {
      int m  = tid >> 4;
      int c2 = (tid & 15) * 2;
      *reinterpret_cast<unsigned*>(&Alds[m][c2]) =
          *reinterpret_cast<const unsigned*>(C1 + ((size_t)(pbase + m) * N_CIN + cb + c2));
    }
    #pragma unroll
    for (int j = 0; j < 2; ++j) {
      int idx = tid + j * 256;
      if (idx < NHEAD * 4) {
        int row = idx >> 2;
        int c8  = (idx & 3) * 8;
        *reinterpret_cast<uint4*>(&Blds[row][c8]) =
            *reinterpret_cast<const uint4*>(Wh + ((size_t)row * 512 + cb + c8));
      }
    }
    __syncthreads();
    if (wave < 6) {
      v16bf a, bb;
      int m  = lane & 15;
      int k0 = (lane < 16) ? 0 : 8;
      #pragma unroll
      for (int i = 0; i < 8; ++i) {
        a[i]     = Alds[m][k0 + i];
        a[8 + i] = Alds[m][16 + k0 + i];
      }
      int nrow = wave * 16 + (lane & 15);
      int kb   = (lane < 16) ? 0 : 8;
      #pragma unroll
      for (int i = 0; i < 8; ++i) { bb[i] = Blds[nrow][kb + i]; bb[8 + i] = Blds[nrow][16 + kb + i]; }
      acc = __builtin_amdgcn_wmma_f32_16x16x32_bf16(false, a, false, bb,
                                                    (short)0, acc, false, false);
    }
    __syncthreads();
  }
  if (wave < 6) {
    const int co = wave * 16 + (lane & 15);
    const float bsv = biasH[co];
    const int mh = (lane < 16) ? 0 : 8;
    #pragma unroll
    for (int r = 0; r < 8; ++r)
      S[(size_t)(pbase + r + mh) * NHEAD + co] = acc[r] + bsv;
  }
}

// ============================================================
// K4: per-pixel epilogue: softmaxes -> d_out, proposals -> ws
// ============================================================
__global__ __launch_bounds__(256) void k_epilogue(const float* __restrict__ S,
                                                  const float* __restrict__ im_info,
                                                  float* __restrict__ out,
                                                  float* __restrict__ P,
                                                  float* __restrict__ Sc,
                                                  float* __restrict__ Ar) {
  int p = blockIdx.x * 256 + threadIdx.x;
  if (p >= N_PIX) return;
  const int b  = p >> 12;
  const int sp = p & 4095;           // y*64 + x
  const int y  = sp >> 6, x = sp & 63;
  const float* s = S + (size_t)p * NHEAD;

  float fg[9];
  #pragma unroll
  for (int a = 0; a < 9; ++a) {
    float s0 = s[a], s1 = s[9 + a];
    float m = fmaxf(s0, s1);
    float e0 = __expf(s0 - m), e1 = __expf(s1 - m);
    float inv = 1.f / (e0 + e1);
    out[OFF_CLS + (((size_t)b * 18 + a) << 12) + sp]     = e0 * inv;
    out[OFF_CLS + (((size_t)b * 18 + 9 + a) << 12) + sp] = e1 * inv;
    fg[a] = e1 * inv;
  }
  #pragma unroll
  for (int c = 0; c < 4; ++c) {
    float s0 = s[54 + c], s1 = s[54 + 4 + c];
    float m = fmaxf(s0, s1);
    float e0 = __expf(s0 - m), e1 = __expf(s1 - m);
    float inv = 1.f / (e0 + e1);
    out[OFF_DW + (((size_t)b * 8 + c) << 12) + sp]     = e0 * inv;
    out[OFF_DW + (((size_t)b * 8 + 4 + c) << 12) + sp] = e1 * inv;
    s0 = s[62 + c]; s1 = s[62 + 4 + c];
    m = fmaxf(s0, s1);
    e0 = __expf(s0 - m); e1 = __expf(s1 - m);
    inv = 1.f / (e0 + e1);
    out[OFF_DH + (((size_t)b * 8 + c) << 12) + sp]     = e0 * inv;
    out[OFF_DH + (((size_t)b * 8 + 4 + c) << 12) + sp] = e1 * inv;
  }
  #pragma unroll
  for (int c = 0; c < 8; ++c) {
    out[OFF_BW + (((size_t)b * 8 + c) << 12) + sp] = s[70 + c];
    out[OFF_BH + (((size_t)b * 8 + c) << 12) + sp] = s[78 + c];
  }
  const float AW[3] = {23.f, 16.f, 11.f};
  const float AH[3] = {12.f, 16.f, 22.f};
  const float SCL[3] = {8.f, 16.f, 32.f};
  const float im_h = im_info[b * 3 + 0];
  const float im_w = im_info[b * 3 + 1];
  const float cx = (float)x * 16.f + 8.f;
  const float cy = (float)y * 16.f + 8.f;
  #pragma unroll
  for (int a = 0; a < 9; ++a) {
    const int ri = a / 3, si = a % 3;
    const float aw = AW[ri] * SCL[si];
    const float ah = AH[ri] * SCL[si];
    const float dxv = s[18 + a * 4 + 0];
    const float dyv = s[18 + a * 4 + 1];
    const float dwv = s[18 + a * 4 + 2];
    const float dhv = s[18 + a * 4 + 3];
    const float pcx = dxv * aw + cx;
    const float pcy = dyv * ah + cy;
    const float pw  = __expf(dwv) * aw;
    const float ph  = __expf(dhv) * ah;
    float x1 = pcx - 0.5f * pw, y1 = pcy - 0.5f * ph;
    float x2 = pcx + 0.5f * pw, y2 = pcy + 0.5f * ph;
    x1 = fminf(fmaxf(x1, 0.f), im_w - 1.f);
    x2 = fminf(fmaxf(x2, 0.f), im_w - 1.f);
    y1 = fminf(fmaxf(y1, 0.f), im_h - 1.f);
    y2 = fminf(fmaxf(y2, 0.f), im_h - 1.f);
    const size_t n = (size_t)b * N_ALL + (size_t)sp * 9 + a;
    P[n * 4 + 0] = x1; P[n * 4 + 1] = y1; P[n * 4 + 2] = x2; P[n * 4 + 3] = y2;
    Sc[n] = fg[a];
    Ar[n] = (x2 - x1 + 1.f) * (y2 - y1 + 1.f);
  }
}

// ============================================================
// K5: top-6000 gate via histogram threshold (deterministic)
// ============================================================
__global__ __launch_bounds__(1024) void k_topk_gate(const float* __restrict__ Sc,
                                                    float* __restrict__ live) {
  __shared__ int hist[2048];
  __shared__ int sbstar, squota;
  const int b = blockIdx.x, tid = threadIdx.x;
  const float* sc = Sc + (size_t)b * N_ALL;
  float* lv = live + (size_t)b * N_ALL;
  for (int i = tid; i < 2048; i += 1024) hist[i] = 0;
  __syncthreads();
  for (int n = tid; n < N_ALL; n += 1024) {
    int bin = (int)(sc[n] * 2048.f);
    bin = bin < 0 ? 0 : (bin > 2047 ? 2047 : bin);
    atomicAdd(&hist[bin], 1);
  }
  __syncthreads();
  if (tid == 0) {
    int cum = 0, bstar = 0;
    for (int bin = 2047; bin >= 0; --bin) {
      cum += hist[bin];
      if (cum >= PRE_TOPN) { bstar = bin; break; }
    }
    sbstar = bstar;
    squota = PRE_TOPN - (cum - hist[bstar]);
  }
  __syncthreads();
  const int bstar = sbstar;
  for (int n = tid; n < N_ALL; n += 1024) {
    float s = sc[n];
    int bin = (int)(s * 2048.f);
    bin = bin < 0 ? 0 : (bin > 2047 ? 2047 : bin);
    lv[n] = (bin > bstar) ? s : -__builtin_inff();
  }
  __syncthreads();
  if (tid == 0) {  // fill boundary bin in index order (deterministic)
    int cnt = 0; const int quota = squota;
    for (int n = 0; n < N_ALL && cnt < quota; ++n) {
      float s = sc[n];
      int bin = (int)(s * 2048.f);
      bin = bin < 0 ? 0 : (bin > 2047 ? 2047 : bin);
      if (bin == bstar) { lv[n] = s; ++cnt; }
    }
  }
}

// ============================================================
// K6: greedy NMS, one block per image, 300 iterations
// ============================================================
__global__ __launch_bounds__(1024) void k_nms(const float* __restrict__ P,
                                              const float* __restrict__ Ar,
                                              float* __restrict__ live,
                                              float* __restrict__ out) {
  __shared__ float sval[1024];
  __shared__ int   sidx[1024];
  __shared__ float cbox[5];
  const int b = blockIdx.x, tid = threadIdx.x;
  const float* box = P + (size_t)b * N_ALL * 4;
  const float* ar  = Ar + (size_t)b * N_ALL;
  float* lv = live + (size_t)b * N_ALL;

  for (int it = 0; it < POST_N; ++it) {
    float v = -__builtin_inff(); int id = 0;
    for (int n = tid; n < N_ALL; n += 1024) {
      float s = lv[n];
      if (s > v) { v = s; id = n; }
    }
    sval[tid] = v; sidx[tid] = id;
    __syncthreads();
    for (int off = 512; off > 0; off >>= 1) {
      if (tid < off) {
        float v2 = sval[tid + off]; int i2 = sidx[tid + off];
        if (v2 > sval[tid] || (v2 == sval[tid] && i2 < sidx[tid])) {
          sval[tid] = v2; sidx[tid] = i2;
        }
      }
      __syncthreads();
    }
    if (tid == 0) {
      const int k = sidx[0];
      const float x1 = box[k * 4 + 0], y1 = box[k * 4 + 1];
      const float x2 = box[k * 4 + 2], y2 = box[k * 4 + 3];
      float* o = out + OFF_ROIS + ((size_t)b * POST_N + it) * 5;
      o[0] = (float)b; o[1] = x1; o[2] = y1; o[3] = x2; o[4] = y2;
      cbox[0] = x1; cbox[1] = y1; cbox[2] = x2; cbox[3] = y2; cbox[4] = ar[k];
    }
    __syncthreads();
    const float bx1 = cbox[0], by1 = cbox[1], bx2 = cbox[2], by2 = cbox[3], ba = cbox[4];
    for (int n = tid; n < N_ALL; n += 1024) {
      if (lv[n] != -__builtin_inff()) {
        float xx1 = fmaxf(bx1, box[n * 4 + 0]);
        float yy1 = fmaxf(by1, box[n * 4 + 1]);
        float xx2 = fminf(bx2, box[n * 4 + 2]);
        float yy2 = fminf(by2, box[n * 4 + 3]);
        float inter = fmaxf(xx2 - xx1 + 1.f, 0.f) * fmaxf(yy2 - yy1 + 1.f, 0.f);
        float iou = inter / (ba + ar[n] - inter);
        if (iou > NMS_TH) lv[n] = -__builtin_inff();
      }
    }
    __syncthreads();
  }
}

// ============================================================
extern "C" void kernel_launch(void* const* d_in, const int* in_sizes, int n_in,
                              void* d_out, int out_size, void* d_ws, size_t ws_size,
                              hipStream_t stream) {
  const float* base_feat = (const float*)d_in[0];
  const float* im_info   = (const float*)d_in[1];
  const float* conv_w    = (const float*)d_in[4];
  const float* conv_b    = (const float*)d_in[5];
  const float* cls_w     = (const float*)d_in[6];
  const float* cls_b     = (const float*)d_in[7];
  const float* bbox_w    = (const float*)d_in[8];
  const float* bbox_b    = (const float*)d_in[9];
  const float* dw_w      = (const float*)d_in[10];
  const float* dw_b      = (const float*)d_in[11];
  const float* dh_w      = (const float*)d_in[12];
  const float* dh_b      = (const float*)d_in[13];
  const float* bw_w      = (const float*)d_in[14];
  const float* bw_b      = (const float*)d_in[15];
  const float* bh_w      = (const float*)d_in[16];
  const float* bh_b      = (const float*)d_in[17];
  float* out = (float*)d_out;

  // ---- workspace layout (256B aligned slabs) ----
  char* ws = (char*)d_ws;
  size_t off = 0;
  auto carve = [&](size_t bytes) {
    char* p = ws + off;
    off += (bytes + 255) & ~(size_t)255;
    return p;
  };
  __bf16* Fpad  = (__bf16*)carve((size_t)N_B * N_HWP * N_HWP * N_CIN * 2); // 17.8 MB
  __bf16* W2    = (__bf16*)carve((size_t)9 * 512 * 512 * 2);               // 4.7 MB
  __bf16* Wh    = (__bf16*)carve((size_t)NHEAD * 512 * 2);
  float*  biasH = (float*) carve((size_t)NHEAD * 4);
  __bf16* C1    = (__bf16*)carve((size_t)N_PIX * N_CIN * 2);               // 16 MB
  float*  Sraw  = (float*) carve((size_t)N_PIX * NHEAD * 4);               // 6.3 MB
  float*  P     = (float*) carve((size_t)N_B * N_ALL * 4 * 4);
  float*  Sc    = (float*) carve((size_t)N_B * N_ALL * 4);
  float*  Ar    = (float*) carve((size_t)N_B * N_ALL * 4);
  float*  Lv    = (float*) carve((size_t)N_B * N_ALL * 4);
  (void)ws_size; (void)n_in; (void)in_sizes; (void)out_size;

  // K0: convert + pad activations
  {
    size_t n = (size_t)N_B * N_HWP * N_HWP * N_CIN;
    k_nchw_to_nhwc_pad<<<dim3((unsigned)((n + 255) / 256)), dim3(256), 0, stream>>>(base_feat, Fpad);
  }
  // K1: pack weights
  {
    size_t n = (size_t)9 * 512 * 512;
    k_pack_w2<<<dim3((unsigned)((n + 255) / 256)), dim3(256), 0, stream>>>(conv_w, W2);
    int nh = NHEAD * 512 + NHEAD;
    k_pack_head<<<dim3((nh + 255) / 256), dim3(256), 0, stream>>>(
        cls_w, cls_b, bbox_w, bbox_b, dw_w, dw_b, dh_w, dh_b, bw_w, bw_b, bh_w, bh_b,
        Wh, biasH);
  }
  // K2: 3x3 conv + ReLU (WMMA bf16, async-to-LDS double buffer)
  k_conv3x3<<<dim3(N_PIX / 32, N_CIN / 128), dim3(256), 25600, stream>>>(Fpad, W2, conv_b, C1);
  // K3: fused 1x1 heads (WMMA bf16)
  k_head1x1<<<dim3(N_PIX / 16), dim3(256), 0, stream>>>(C1, Wh, biasH, Sraw);
  // K4: softmaxes + proposal decode
  k_epilogue<<<dim3(N_PIX / 256), dim3(256), 0, stream>>>(Sraw, im_info, out, P, Sc, Ar);
  // K5: top-6000 gate
  k_topk_gate<<<dim3(N_B), dim3(1024), 0, stream>>>(Sc, Lv);
  // K6: greedy NMS -> rois
  k_nms<<<dim3(N_B), dim3(1024), 0, stream>>>(P, Ar, Lv, out);
}
